// TrajectoryGenerator_46351287058998
// MI455X (gfx1250) — matrix operand order repaired and checked by reading, not compiled, not run on hardware
//
#include <hip/hip_runtime.h>
#include <hip/hip_bf16.h>

// =====================================================================
// Set-Transformer trajectory generator for MI455X (gfx1250, wave32).
// - All GEMM-class math on V_WMMA_F32_16X16X32_BF16 (bf16 in, f32 acc).
// - GEMM stages A/B tiles to LDS with GLOBAL_LOAD_ASYNC_TO_LDS_B128,
//   double-buffered, synced with S_WAIT_ASYNCCNT + barrier; fragments
//   then come from LDS (ds_load_b128).
// - Attention: scores + probabilities staged in LDS (up to 240KB of the
//   320KB/WGP), softmax in f32, P@V consumes bf16 probabilities from LDS.
// =====================================================================

typedef __bf16 bf16;
typedef __attribute__((ext_vector_type(16))) __bf16 v16bf;
typedef __attribute__((ext_vector_type(8)))  __bf16 v8bf;
typedef __attribute__((ext_vector_type(8)))  float   v8f;

union V16U { v16bf v; v8bf h[2]; };

// ---------------------------------------------------------------------
// CDNA5 async copy: global -> LDS, 16B per lane, tracked by ASYNCcnt.
// VDST operand supplies the per-lane LDS byte address.
// ---------------------------------------------------------------------
__device__ __forceinline__ void async_b128(unsigned lds_byte_off, const void* gaddr)
{
    asm volatile("global_load_async_to_lds_b128 %0, %1, off"
                 :: "v"(lds_byte_off), "v"(gaddr)
                 : "memory");
}
__device__ __forceinline__ void wait_async0()
{
    asm volatile("s_wait_asynccnt 0" ::: "memory");
}
__device__ __forceinline__ unsigned lds_off(const void* p)
{
    // generic LDS pointer = {aperture, lds_byte_offset[31:0]}  (ISA 10.2)
    return (unsigned)(unsigned long long)p;
}

// ---------------------------------------------------------------------
// WMMA fragment loaders (CDNA5 ISA 7.12.2 layouts, wave32)
// A (16x32, row-major, stride ld):
//   lane l: row m=l&15; halves at K = hi*8..hi*8+7 and 16+hi*8..16+hi*8+7
// B (32x16 = rows of Bt[N][K], stride ld):
//   lane l: row n=l&15; halves at K = hi*8..hi*8+7 and 16+hi*8..16+hi*8+7
// ---------------------------------------------------------------------
__device__ __forceinline__ v16bf frag_a(const bf16* A, int ld, int lane)
{
    int m  = lane & 15;
    int hi = (lane >> 4) & 1;
    const bf16* p = A + (size_t)m * ld + hi * 8;
    V16U u;
    u.h[0] = *(const v8bf*)(p);
    u.h[1] = *(const v8bf*)(p + 16);
    return u.v;
}

__device__ __forceinline__ v16bf frag_b(const bf16* Bt, int ld, int lane)
{
    int n  = lane & 15;
    int hi = (lane >> 4) & 1;
    const bf16* p = Bt + (size_t)n * ld + hi * 8;
    V16U u;
    u.h[0] = *(const v8bf*)(p);
    u.h[1] = *(const v8bf*)(p + 16);
    return u.v;
}

__device__ __forceinline__ v8f wmma_bf16(v16bf a, v16bf b, v8f c)
{
    return __builtin_amdgcn_wmma_f32_16x16x32_bf16(
        false, a, false, b, (short)0, c, false, false);
}

// ---------------------------------------------------------------------
// Weight prep: W[K][N] f32  ->  Wt[Npad][K] bf16 (zero-padded rows)
// ---------------------------------------------------------------------
__global__ void wprep_kernel(const float* __restrict__ W, bf16* __restrict__ Wt,
                             int K, int N, int Npad)
{
    int idx = blockIdx.x * 256 + threadIdx.x;
    if (idx >= Npad * K) return;
    int n = idx / K, k = idx % K;
    float v = (n < N) ? W[(size_t)k * N + n] : 0.f;
    Wt[idx] = (bf16)v;
}

__global__ void zero_bf_kernel(bf16* __restrict__ p, int n)
{
    int i = blockIdx.x * 256 + threadIdx.x;
    if (i < n) p[i] = (bf16)0.f;
}

// ---------------------------------------------------------------------
// Embedding: token m = (group g, agent n, time t); y = x @ embed_w + b
// obs_traj_rel is [20][2048][2]; token r within group = n*20 + t.
// ---------------------------------------------------------------------
__global__ void embed_kernel(const float* __restrict__ rel,
                             const float* __restrict__ ew,
                             const float* __restrict__ eb,
                             float* __restrict__ yf, bf16* __restrict__ ybf)
{
    int m = blockIdx.x;      // 0..40959
    int h = threadIdx.x;     // 0..255
    int g = m / 640, r = m % 640;
    int n = r / 20, t = r % 20;
    int a = g * 32 + n;
    float x0 = rel[(size_t)(t * 2048 + a) * 2 + 0];
    float x1 = rel[(size_t)(t * 2048 + a) * 2 + 1];
    float v  = x0 * ew[h] + x1 * ew[256 + h] + eb[h];
    size_t o = (size_t)m * 256 + h;
    yf[o]  = v;
    ybf[o] = (bf16)v;
}

__global__ void seed_expand_kernel(const float* __restrict__ seed,
                                   float* __restrict__ Sf, bf16* __restrict__ Sbf)
{
    int idx = blockIdx.x * 256 + threadIdx.x;   // 192*256
    int s = (idx >> 8) % 3;
    int h = idx & 255;
    float v = seed[s * 256 + h];
    Sf[idx]  = v;
    Sbf[idx] = (bf16)v;
}

// ---------------------------------------------------------------------
// Generic WMMA GEMM: C[M][N] = act(A[M][K](bf16) @ Wt[Npad][K]^T + bias)
// 8 waves/block; wave (wr,wc) owns rows wr*16, cols wc*32 (2 accumulators).
// A and B 64x32 tiles staged to LDS with async-to-LDS, double buffered;
// WMMA fragments come from LDS.
// act: 0 none, 1 relu, 2 resid + relu (MAB epilogue)
// bfmode: 0 plain [M][N], 1 head-split [B][4][Lpad][64],
//         2 head-split transposed [B][4][64][Lpad]
// ---------------------------------------------------------------------
__global__ __launch_bounds__(256)
void gemm_bf16_kernel(const bf16* __restrict__ A, const bf16* __restrict__ Wt,
                      const float* __restrict__ bias,
                      float* __restrict__ outf, bf16* __restrict__ outbf,
                      const float* __restrict__ resid,
                      int M, int K, int N,
                      int act, int bfmode, int L, int Lpad)
{
    __shared__ bf16 sA[2][64 * 32];
    __shared__ bf16 sB[2][64 * 32];

    int tid  = threadIdx.x;
    int lane = tid & 31, wid = tid >> 5;
    int wr = wid & 3, wc = wid >> 2;
    int hi = lane >> 4, nl = lane & 15;

    int mb = blockIdx.x * 64;
    int nb = blockIdx.y * 64;

    // staging: thread t moves one 16B chunk per tile: row=t/4, seg=(t%4)*8
    int srow = tid >> 2;
    int sseg = (tid & 3) * 8;
    const bf16* gA = A  + (size_t)(mb + srow) * K + sseg;
    const bf16* gB = Wt + (size_t)(nb + srow) * K + sseg;
    unsigned oA0 = lds_off(&sA[0][srow * 32 + sseg]);
    unsigned oA1 = lds_off(&sA[1][srow * 32 + sseg]);
    unsigned oB0 = lds_off(&sB[0][srow * 32 + sseg]);
    unsigned oB1 = lds_off(&sB[1][srow * 32 + sseg]);

    // prologue: stage k-step 0 into buffer 0
    async_b128(oA0, gA);
    async_b128(oB0, gB);

    v8f c0 = {}, c1 = {};
    int nk = K >> 5;
    for (int ks = 0; ks < nk; ++ks) {
        int cur = ks & 1;
        wait_async0();        // my stage(ks) writes have landed in LDS
        __syncthreads();      // everyone's stage(ks) visible; buf[!cur] free
        if (ks + 1 < nk) {    // overlap stage(ks+1) with compute(ks)
            int k1 = (ks + 1) << 5;
            async_b128(cur ? oA0 : oA1, gA + k1);
            async_b128(cur ? oB0 : oB1, gB + k1);
        }
        const bf16* pa  = &sA[cur][(wr * 16) * 32];
        const bf16* pb0 = &sB[cur][(wc * 32) * 32];
        const bf16* pb1 = &sB[cur][(wc * 32 + 16) * 32];
        v16bf a  = frag_a(pa,  32, lane);
        v16bf b0 = frag_b(pb0, 32, lane);
        v16bf b1 = frag_b(pb1, 32, lane);
        c0 = wmma_bf16(a, b0, c0);
        c1 = wmma_bf16(a, b1, c1);
    }

    int row0 = mb + wr * 16;
    int n0   = nb + wc * 32;
    for (int half = 0; half < 2; ++half) {
        v8f cc = half ? c1 : c0;
        int col = n0 + half * 16 + nl;
        if (col >= N) continue;
        float bv = bias ? bias[col] : 0.f;
#pragma unroll
        for (int r = 0; r < 8; ++r) {
            int row = row0 + r + hi * 8;
            float v = cc[r] + bv;
            if (act == 1)      v = fmaxf(v, 0.f);
            else if (act == 2) v = resid[(size_t)row * N + col] + fmaxf(v, 0.f);
            if (outf) outf[(size_t)row * N + col] = v;
            if (outbf) {
                if (bfmode == 0) {
                    outbf[(size_t)row * N + col] = (bf16)v;
                } else {
                    int b = row / L, l = row % L;
                    int h = col >> 6, d = col & 63;
                    size_t idx = (bfmode == 1)
                        ? ((size_t)((b * 4 + h) * Lpad + l) * 64 + d)
                        : ((size_t)((b * 4 + h) * 64 + d) * Lpad + l);
                    outbf[idx] = (bf16)v;
                }
            }
        }
    }
}

// ---------------------------------------------------------------------
// Batched MHA core: out[b][l][h*64+d] (+resid) = softmax(Q Kh^T * scale) @ V
// Qh [B][4][Lqpad][64], Kh [B][4][Lkpad][64], Vt [B][4][64][Lkpad] (bf16).
// One block per (b, h, 64-query tile). S (f32) and P (bf16) live in LDS.
// ---------------------------------------------------------------------
__global__ __launch_bounds__(256)
void attention_kernel(const bf16* __restrict__ Qh, const bf16* __restrict__ Kh,
                      const bf16* __restrict__ Vt,
                      const float* __restrict__ resid,
                      float* __restrict__ outf, bf16* __restrict__ outbf,
                      int Lq, int Lqpad, int Lk, int Lkpad, float scale)
{
    extern __shared__ char smem[];
    float* S = (float*)smem;
    bf16*  P = (bf16*)(smem + (size_t)64 * Lkpad * sizeof(float));

    int nQt = Lqpad >> 6;
    int bx  = blockIdx.x;
    int qt  = bx % nQt;
    int h   = (bx / nQt) & 3;
    int b   = bx / (nQt * 4);
    size_t bh = (size_t)(b * 4 + h);

    int tid  = threadIdx.x;
    int lane = tid & 31, wid = tid >> 5;
    int wr = wid & 3, wc = wid >> 2;
    int hi = lane >> 4, nl = lane & 15;

    const bf16* Qbase = Qh + (bh * Lqpad + qt * 64 + wr * 16) * 64;

    // ---- phase 1: S = Q @ Kh^T * scale -> LDS (masked past Lk) ----
    for (int nt = 0; nt < Lkpad; nt += 64) {
        const bf16* K0 = Kh + (bh * Lkpad + nt + wc * 32) * 64;
        const bf16* K1 = K0 + 16 * 64;
        v8f c0 = {}, c1 = {};
#pragma unroll
        for (int k0 = 0; k0 < 64; k0 += 32) {
            v16bf a  = frag_a(Qbase + k0, 64, lane);
            v16bf b0 = frag_b(K0 + k0, 64, lane);
            v16bf b1 = frag_b(K1 + k0, 64, lane);
            c0 = wmma_bf16(a, b0, c0);
            c1 = wmma_bf16(a, b1, c1);
        }
        for (int half = 0; half < 2; ++half) {
            v8f cc = half ? c1 : c0;
            int col = nt + wc * 32 + half * 16 + nl;
            bool keep = (col < Lk);
#pragma unroll
            for (int r = 0; r < 8; ++r) {
                int rl = wr * 16 + r + hi * 8;
                S[rl * Lkpad + col] = keep ? cc[r] * scale : -1e30f;
            }
        }
    }
    __syncthreads();

    // ---- phase 2: softmax rows (4 lanes per row), write bf16 P ----
    {
        int row = tid >> 2, part = tid & 3;
        float m = -1e30f;
        for (int c = part; c < Lkpad; c += 4) m = fmaxf(m, S[row * Lkpad + c]);
        m = fmaxf(m, __shfl_xor(m, 1, 32));
        m = fmaxf(m, __shfl_xor(m, 2, 32));
        float s = 0.f;
        for (int c = part; c < Lkpad; c += 4) s += __expf(S[row * Lkpad + c] - m);
        s += __shfl_xor(s, 1, 32);
        s += __shfl_xor(s, 2, 32);
        float inv = 1.f / s;
        for (int c = part; c < Lkpad; c += 4)
            P[row * Lkpad + c] = (bf16)(__expf(S[row * Lkpad + c] - m) * inv);
    }
    __syncthreads();

    // ---- phase 3: O = P @ V (A from LDS, B = Vt from global) ----
    const bf16* Pb = P  + (size_t)(wr * 16) * Lkpad;
    const bf16* V0 = Vt + (bh * 64 + wc * 32) * (size_t)Lkpad;
    const bf16* V1 = V0 + (size_t)16 * Lkpad;
    v8f c0 = {}, c1 = {};
    for (int k0 = 0; k0 < Lkpad; k0 += 32) {
        v16bf a  = frag_a(Pb + k0, Lkpad, lane);
        v16bf b0 = frag_b(V0 + k0, Lkpad, lane);
        v16bf b1 = frag_b(V1 + k0, Lkpad, lane);
        c0 = wmma_bf16(a, b0, c0);
        c1 = wmma_bf16(a, b1, c1);
    }
    for (int half = 0; half < 2; ++half) {
        v8f cc = half ? c1 : c0;
        int d = wc * 32 + half * 16 + nl;
#pragma unroll
        for (int r = 0; r < 8; ++r) {
            int l = qt * 64 + wr * 16 + r + hi * 8;
            if (l >= Lq) continue;
            size_t o = ((size_t)(b * Lq + l)) * 256 + h * 64 + d;
            float v = cc[r];
            if (resid) v += resid[o];
            if (outf)  outf[o]  = v;
            if (outbf) outbf[o] = (bf16)v;
        }
    }
}

// ---------------------------------------------------------------------
// Fused residual-add + LayerNorm over H=256: one wave per row.
// ---------------------------------------------------------------------
__global__ __launch_bounds__(256)
void ln_kernel(const float* __restrict__ a, const float* __restrict__ bres,
               const float* __restrict__ g, const float* __restrict__ beta,
               float* __restrict__ outf, bf16* __restrict__ outbf)
{
    int wid = threadIdx.x >> 5, lane = threadIdx.x & 31;
    size_t row = (size_t)blockIdx.x * 8 + wid;
    const float* pa = a    + row * 256;
    const float* pb = bres + row * 256;
    float x[8];
    float s = 0.f;
#pragma unroll
    for (int i = 0; i < 8; ++i) {
        int c = lane + i * 32;
        x[i] = pa[c] + pb[c];
        s += x[i];
    }
    for (int off = 1; off < 32; off <<= 1) s += __shfl_xor(s, off, 32);
    float mean = s * (1.f / 256.f);
    float v = 0.f;
#pragma unroll
    for (int i = 0; i < 8; ++i) { float d = x[i] - mean; v += d * d; }
    for (int off = 1; off < 32; off <<= 1) v += __shfl_xor(v, off, 32);
    float inv = rsqrtf(v * (1.f / 256.f) + 1e-5f);
#pragma unroll
    for (int i = 0; i < 8; ++i) {
        int c = lane + i * 32;
        float o = (x[i] - mean) * inv * g[c] + beta[c];
        outf[row * 256 + c]  = o;
        outbf[row * 256 + c] = (bf16)o;
    }
}

// =====================================================================
// Host orchestration
// =====================================================================
extern "C" void kernel_launch(void* const* d_in, const int* in_sizes, int n_in,
                              void* d_out, int out_size, void* d_ws, size_t ws_size,
                              hipStream_t stream)
{
    (void)in_sizes; (void)n_in; (void)out_size; (void)ws_size;

    // Input order: JAX sorted-key pytree flatten of setup_inputs() dict:
    //  0 obs_traj, 1 obs_traj_rel,
    //  2 embed_b, 3 embed_w,
    //  4..19 enc[0]: b1,b2,bk,bo,bq,bv,ln1_b,ln1_g,ln2_b,ln2_g,w1,w2,wk,wo,wq,wv
    //  20..35 enc[1] (same order),
    //  36..43 pma: bk,bo,bq,bv,wk,wo,wq,wv
    //  44 pma_seed, 45 reg_b, 46 reg_w,
    //  47..54 sab: bk,bo,bq,bv,wk,wo,wq,wv,
    //  55 start_end_seq (unused: equal-size contiguous groups)
    auto F = [&](int i) { return (const float*)d_in[i]; };
    const float* rel = F(1);
    const int ENC0 = 4, PMA = 36, SEED = 44, REGB = 45, REGW = 46, SAB = 47;
    const int eB1 = 0, eB2 = 1, eBK = 2, eBO = 3, eBQ = 4, eBV = 5,
              eLN1B = 6, eLN1G = 7, eLN2B = 8, eLN2G = 9,
              eW1 = 10, eW2 = 11, eWK = 12, eWO = 13, eWQ = 14, eWV = 15;
    const int aBK = 0, aBO = 1, aBQ = 2, aBV = 3, aWK = 4, aWO = 5, aWQ = 6, aWV = 7;

    const int M_ENC = 64 * 640;   // 40960 tokens

    // ---- workspace carve (bump allocator) ----
    char* ws = (char*)d_ws;
    size_t off = 0;
    auto takeb = [&](size_t bytes) -> void* {
        void* p = (void*)(ws + off);
        off += (bytes + 255) & ~(size_t)255;
        return p;
    };
    float* yf  = (float*)takeb((size_t)M_ENC * 256 * 4);
    bf16*  ybf = (bf16*) takeb((size_t)M_ENC * 256 * 2);
    bf16*  Qh  = (bf16*) takeb((size_t)64 * 4 * 640 * 64 * 2);
    bf16*  Kh  = (bf16*) takeb((size_t)64 * 4 * 640 * 64 * 2);
    bf16*  Vt  = (bf16*) takeb((size_t)64 * 4 * 640 * 64 * 2);
    bf16*  abf = (bf16*) takeb((size_t)M_ENC * 512 * 2);   // attn-out / FFN hidden
    float* pf  = (float*)takeb((size_t)M_ENC * 256 * 4);   // proj / FFN out
    bf16*  QhD = (bf16*) takeb((size_t)64 * 4 * 64 * 64 * 2);
    bf16*  KhD = (bf16*) takeb((size_t)64 * 4 * 64 * 64 * 2);
    bf16*  VtD = (bf16*) takeb((size_t)64 * 4 * 64 * 64 * 2);
    float* Sf  = (float*)takeb((size_t)192 * 256 * 4);
    bf16*  Sbf = (bf16*) takeb((size_t)192 * 256 * 2);
    float* Qdf = (float*)takeb((size_t)192 * 256 * 4);
    float* Of  = (float*)takeb((size_t)192 * 256 * 4);
    bf16*  Obf = (bf16*) takeb((size_t)192 * 256 * 2);
    float* O2f = (float*)takeb((size_t)192 * 256 * 4);
    bf16*  O2bf= (bf16*) takeb((size_t)192 * 256 * 2);
    float* Q2f = (float*)takeb((size_t)192 * 256 * 4);
    float* O3f = (float*)takeb((size_t)192 * 256 * 4);
    bf16*  O3bf= (bf16*) takeb((size_t)192 * 256 * 2);
    bf16*  O4bf= (bf16*) takeb((size_t)192 * 256 * 2);

    // ---- weight prep: f32 [K][N] -> bf16 [Npad][K] ----
    auto prep = [&](int idx, int K, int N, int Npad) -> bf16* {
        bf16* dst = (bf16*)takeb((size_t)Npad * K * 2);
        int total = Npad * K;
        wprep_kernel<<<dim3((total + 255) / 256), dim3(256), 0, stream>>>(
            F(idx), dst, K, N, Npad);
        return dst;
    };
    bf16 *ewq[2], *ewk[2], *ewv[2], *ewo[2], *ew1[2], *ew2[2];
    for (int blk = 0; blk < 2; ++blk) {
        int base = ENC0 + blk * 16;
        ewq[blk] = prep(base + eWQ, 256, 256, 256);
        ewk[blk] = prep(base + eWK, 256, 256, 256);
        ewv[blk] = prep(base + eWV, 256, 256, 256);
        ewo[blk] = prep(base + eWO, 256, 256, 256);
        ew1[blk] = prep(base + eW1, 256, 512, 512);
        ew2[blk] = prep(base + eW2, 512, 256, 256);
    }
    bf16* pwq = prep(PMA + aWQ, 256, 256, 256);
    bf16* pwk = prep(PMA + aWK, 256, 256, 256);
    bf16* pwv = prep(PMA + aWV, 256, 256, 256);
    bf16* pwo = prep(PMA + aWO, 256, 256, 256);
    bf16* swq = prep(SAB + aWQ, 256, 256, 256);
    bf16* swk = prep(SAB + aWK, 256, 256, 256);
    bf16* swv = prep(SAB + aWV, 256, 256, 256);
    bf16* swo = prep(SAB + aWO, 256, 256, 256);
    bf16* rwt = prep(REGW, 256, 60, 64);

    // ---- launch helpers ----
    auto gemm = [&](const bf16* A, const bf16* Wt, const float* bias,
                    float* outf, bf16* outbf, const float* resid,
                    int M, int K, int N, int Npad,
                    int act, int bfmode, int L, int Lpad) {
        gemm_bf16_kernel<<<dim3(M / 64, Npad / 64), dim3(256), 0, stream>>>(
            A, Wt, bias, outf, outbf, resid, M, K, N, act, bfmode, L, Lpad);
    };
    auto attn = [&](const bf16* Q, const bf16* K, const bf16* V,
                    const float* resid, float* of, bf16* ob,
                    int Lq, int Lqpad, int Lk, int Lkpad, float scale) {
        int blocks = 64 * 4 * (Lqpad / 64);
        size_t sm = (size_t)64 * Lkpad * 6;   // f32 S + bf16 P
        attention_kernel<<<dim3(blocks), dim3(256), sm, stream>>>(
            Q, K, V, resid, of, ob, Lq, Lqpad, Lk, Lkpad, scale);
    };
    auto zero = [&](bf16* p, int n) {
        zero_bf_kernel<<<dim3((n + 255) / 256), dim3(256), 0, stream>>>(p, n);
    };

    // ---- embedding ----
    embed_kernel<<<dim3(M_ENC), dim3(256), 0, stream>>>(rel, F(3), F(2), yf, ybf);

    // ---- encoder blocks ----
    for (int blk = 0; blk < 2; ++blk) {
        int base = ENC0 + blk * 16;
        gemm(ybf, ewq[blk], F(base + eBQ), nullptr, Qh, nullptr,
             M_ENC, 256, 256, 256, 0, 1, 640, 640);
        gemm(ybf, ewk[blk], F(base + eBK), nullptr, Kh, nullptr,
             M_ENC, 256, 256, 256, 0, 1, 640, 640);
        gemm(ybf, ewv[blk], F(base + eBV), nullptr, Vt, nullptr,
             M_ENC, 256, 256, 256, 0, 2, 640, 640);
        attn(Qh, Kh, Vt, nullptr, nullptr, abf, 640, 640, 640, 640, 0.125f);
        gemm(abf, ewo[blk], F(base + eBO), pf, nullptr, nullptr,
             M_ENC, 256, 256, 256, 0, 0, 640, 640);
        ln_kernel<<<dim3(M_ENC / 8), dim3(256), 0, stream>>>(
            pf, yf, F(base + eLN1G), F(base + eLN1B), yf, ybf);
        gemm(ybf, ew1[blk], F(base + eB1), nullptr, abf, nullptr,
             M_ENC, 256, 512, 512, 1, 0, 640, 640);
        gemm(abf, ew2[blk], F(base + eB2), pf, nullptr, nullptr,
             M_ENC, 512, 256, 256, 0, 0, 640, 640);
        ln_kernel<<<dim3(M_ENC / 8), dim3(256), 0, stream>>>(
            pf, yf, F(base + eLN2G), F(base + eLN2B), yf, ybf);
    }

    // ---- decoder: PMA then SAB (MAB, scale = 1/sqrt(H) = 1/16) ----
    seed_expand_kernel<<<dim3(192), dim3(256), 0, stream>>>(F(SEED), Sf, Sbf);
    zero(QhD, 64 * 4 * 64 * 64);
    zero(KhD, 64 * 4 * 64 * 64);
    zero(VtD, 64 * 4 * 64 * 64);

    // PMA: Q from seeds (Lq=3 pad 64), K/V from encoder output y
    gemm(Sbf, pwq, F(PMA + aBQ), Qdf, QhD, nullptr,
         192, 256, 256, 256, 0, 1, 3, 64);
    gemm(ybf, pwk, F(PMA + aBK), nullptr, Kh, nullptr,
         M_ENC, 256, 256, 256, 0, 1, 640, 640);
    gemm(ybf, pwv, F(PMA + aBV), nullptr, Vt, nullptr,
         M_ENC, 256, 256, 256, 0, 2, 640, 640);
    attn(QhD, Kh, Vt, Qdf, Of, Obf, 3, 64, 640, 640, 0.0625f);
    gemm(Obf, pwo, F(PMA + aBO), O2f, O2bf, Of,
         192, 256, 256, 256, 2, 0, 3, 64);

    // SAB: self-attention on the 3 seeds (Lk=3 pad 64)
    gemm(O2bf, swq, F(SAB + aBQ), Q2f, QhD, nullptr,
         192, 256, 256, 256, 0, 1, 3, 64);
    gemm(O2bf, swk, F(SAB + aBK), nullptr, KhD, nullptr,
         192, 256, 256, 256, 0, 1, 3, 64);
    gemm(O2bf, swv, F(SAB + aBV), nullptr, VtD, nullptr,
         192, 256, 256, 256, 0, 2, 3, 64);
    attn(QhD, KhD, VtD, Q2f, O3f, O3bf, 3, 64, 3, 64, 0.0625f);
    gemm(O3bf, swo, F(SAB + aBO), nullptr, O4bf, O3f,
         192, 256, 256, 256, 2, 0, 3, 64);

    // ---- regressor: [192,256] @ [256,60] + b -> d_out [192,30,2] ----
    gemm(O4bf, rwt, F(REGB), (float*)d_out, nullptr, nullptr,
         192, 256, 60, 64, 0, 0, 3, 64);
}
